// GANTGNN_13005160972789
// MI455X (gfx1250) — matrix operand order repaired and checked by reading, not compiled, not run on hardware
//
#include <hip/hip_runtime.h>
#include <hip/hip_bf16.h>
#include <stdint.h>

typedef __attribute__((ext_vector_type(2))) float v2f;
typedef __attribute__((ext_vector_type(8))) float v8f;
typedef __attribute__((ext_vector_type(16))) _Float16 v16h;
typedef __attribute__((ext_vector_type(4))) int v4i;
typedef __attribute__((address_space(1))) v4i* gv4p;  // global int4*
typedef __attribute__((address_space(3))) v4i* lv4p;  // LDS int4*

#define GAT_ALPHA 0.2f
#define N_NODES 4096
#define IN_FEAT 256
#define OUT_FEAT 256
#define HEADS 4
#define HEAD_DIM 64
#define LN_EPS 1e-5f

#define JB 128          // adj columns staged per iteration in pass B
#define NB (N_NODES / JB)
#define ADJ_STRIDE 132  // 128 + 4 pad: row stride % 64 banks == 4 -> conflict-free

#if defined(__has_builtin)
#if __has_builtin(__builtin_amdgcn_global_load_async_to_lds_b128) && \
    __has_builtin(__builtin_amdgcn_s_wait_asynccnt)
#define GAT_ASYNC 1
#endif
#endif
#ifndef GAT_ASYNC
#define GAT_ASYNC 0
#endif

// ---------------------------------------------------------------------------
// Kernel 1: h = x @ W  (fp32 WMMA 16x16x4 -- keep full precision for logits)
// grid (N/16, 4), block 128 = 4 waves; wave w owns col tile (blockIdx.y*4 + w)
// ---------------------------------------------------------------------------
__global__ __launch_bounds__(128) void gat_gemm_xw(const float* __restrict__ x,
                                                   const float* __restrict__ W,
                                                   float* __restrict__ h) {
    const int lane = threadIdx.x & 31;
    const int wave = threadIdx.x >> 5;
    const int i0 = blockIdx.x * 16;
    const int c0 = (blockIdx.y * 4 + wave) * 16;
    const int mr = lane & 15;          // M row within tile (A) / N col (B,C)
    const int kh = (lane >> 4) << 1;   // 0 for lanes 0-15, 2 for lanes 16-31

    v8f acc = {};
    const float* xrow = x + (size_t)(i0 + mr) * IN_FEAT;
    for (int k = 0; k < IN_FEAT; k += 4) {
        v2f a;
        a.x = xrow[k + kh];
        a.y = xrow[k + kh + 1];
        v2f b;
        b.x = W[(size_t)(k + kh) * OUT_FEAT + c0 + mr];
        b.y = W[(size_t)(k + kh + 1) * OUT_FEAT + c0 + mr];
        acc = __builtin_amdgcn_wmma_f32_16x16x4_f32(false, a, false, b,
                                                    (short)0, acc, false, false);
    }
    const int rbase = i0 + ((lane >> 4) << 3);  // lanes 16-31 hold rows M+8
#pragma unroll
    for (int r = 0; r < 8; ++r)
        h[(size_t)(rbase + r) * OUT_FEAT + c0 + mr] = acc[r];
}

// ---------------------------------------------------------------------------
// Kernel 2: src[n,h] = h[n,h,:] . a[:64] ; dst[n,h] = h[n,h,:] . a[64:]
// ---------------------------------------------------------------------------
__global__ __launch_bounds__(256) void gat_srcdst(const float* __restrict__ h,
                                                  const float* __restrict__ avec,
                                                  float* __restrict__ src,
                                                  float* __restrict__ dst) {
    const int p = blockIdx.x * 256 + threadIdx.x;  // (node, head) pair
    if (p >= N_NODES * HEADS) return;
    const int i = p >> 2, hh = p & 3;
    const float* hp = h + (size_t)i * OUT_FEAT + hh * HEAD_DIM;
    float s = 0.f, d = 0.f;
#pragma unroll 8
    for (int k = 0; k < HEAD_DIM; ++k) {
        const float v = hp[k];
        s += v * avec[k];
        d += v * avec[HEAD_DIM + k];
    }
    src[p] = s;
    dst[p] = d;
}

// ---------------------------------------------------------------------------
// Kernel 3: transpose + convert h[4096][256] f32 -> hT[256][4096] f16
// ---------------------------------------------------------------------------
__global__ __launch_bounds__(256) void gat_h2t(const float* __restrict__ h,
                                               _Float16* __restrict__ hT) {
    __shared__ float tile[32][33];
    const int n0 = blockIdx.x * 32;
    const int c0 = blockIdx.y * 32;
    const int tx = threadIdx.x & 31;
    const int ty = threadIdx.x >> 5;  // 0..7
#pragma unroll
    for (int r = ty; r < 32; r += 8)
        tile[r][tx] = h[(size_t)(n0 + r) * OUT_FEAT + c0 + tx];
    __syncthreads();
#pragma unroll
    for (int r = ty; r < 32; r += 8)
        hT[(size_t)(c0 + r) * N_NODES + n0 + tx] = (_Float16)tile[tx][r];
}

// ---------------------------------------------------------------------------
// Kernel 4: per-row masked softmax stats (online max / sum-exp), one block/row
// ---------------------------------------------------------------------------
__global__ __launch_bounds__(256) void gat_rowstats(const float* __restrict__ adj,
                                                    const float* __restrict__ src,
                                                    const float* __restrict__ dst,
                                                    float* __restrict__ rowmax,
                                                    float* __restrict__ rowsinv) {
    __shared__ float redM[HEADS][256];
    __shared__ float redS[HEADS][256];
    const int i = blockIdx.x;
    const int t = threadIdx.x;

    float sm[HEADS], m[HEADS], s[HEADS];
#pragma unroll
    for (int hh = 0; hh < HEADS; ++hh) {
        sm[hh] = src[i * HEADS + hh];
        m[hh] = -3.0e38f;
        s[hh] = 0.f;
    }
    const float* arow = adj + (size_t)i * N_NODES;
    for (int j = t; j < N_NODES; j += 256) {
        const float av = arow[j];
        if (av > 0.f) {
#pragma unroll
            for (int hh = 0; hh < HEADS; ++hh) {
                float e = sm[hh] + dst[j * HEADS + hh];
                e = (e > 0.f) ? e : GAT_ALPHA * e;
                if (e > m[hh]) {
                    s[hh] = s[hh] * __expf(m[hh] - e) + 1.f;
                    m[hh] = e;
                } else {
                    s[hh] += __expf(e - m[hh]);
                }
            }
        }
    }
#pragma unroll
    for (int hh = 0; hh < HEADS; ++hh) { redM[hh][t] = m[hh]; redS[hh][t] = s[hh]; }
    __syncthreads();
    for (int st = 128; st > 0; st >>= 1) {
        if (t < st) {
#pragma unroll
            for (int hh = 0; hh < HEADS; ++hh) {
                const float m1 = redM[hh][t], m2 = redM[hh][t + st];
                const float s1 = redS[hh][t], s2 = redS[hh][t + st];
                const float mm = fmaxf(m1, m2);
                redM[hh][t] = mm;
                redS[hh][t] = s1 * __expf(m1 - mm) + s2 * __expf(m2 - mm);
            }
        }
        __syncthreads();
    }
    if (t < HEADS) {
        rowmax[i * HEADS + t] = redM[t][0];
        rowsinv[i * HEADS + t] = 1.0f / redS[t][0];
    }
}

// ---------------------------------------------------------------------------
// Kernel 5: out[i,h,:] = softmax-row . h[:,h,:]  via f16 WMMA 16x16x32,
// double-buffered adj staging through async global->LDS copies (ASYNCcnt).
// grid N/16 blocks, block 128 = 4 waves; wave w = head w, shared adj tile
// ---------------------------------------------------------------------------
__global__ __launch_bounds__(128) void gat_aggregate(const float* __restrict__ adj,
                                                     const _Float16* __restrict__ hT,
                                                     const float* __restrict__ src,
                                                     const float* __restrict__ dst,
                                                     const float* __restrict__ rowmax,
                                                     const float* __restrict__ rowsinv,
                                                     float* __restrict__ out) {
    __shared__ float adjT[2][16][ADJ_STRIDE];
    __shared__ float dstT[2][JB][HEADS];
    const int tid = threadIdx.x;
    const int lane = tid & 31;
    const int head = tid >> 5;
    const int i0 = blockIdx.x * 16;
    const int mr = lane & 15;           // A-matrix M row / B,C N column
    const int kbase = (lane >> 4) * 8;  // unused: A and B both take kb16
    const int kb16 = (lane >> 4) * 16;  // B K offset for this half-wave

    const int myrow = i0 + mr;
    const float srcv = src[myrow * HEADS + head];
    const float mv = rowmax[myrow * HEADS + head];
    const float isv = rowsinv[myrow * HEADS + head];

    // issue staging of adj[i0:i0+16, j0:j0+128] and dst[j0:j0+128, :] into buf
    auto stage = [&](int buf, int j0) {
        const int r0 = tid >> 5;
        const int c = (tid & 31) * 4;
#if GAT_ASYNC
#pragma unroll
        for (int it = 0; it < 4; ++it) {
            const int r = r0 + it * 4;
            const float* gp = adj + (size_t)(i0 + r) * N_NODES + j0 + c;
            float* lp = &adjT[buf][r][c];
            __builtin_amdgcn_global_load_async_to_lds_b128(
                (gv4p)(uintptr_t)gp, (lv4p)(uint32_t)(uintptr_t)lp, 0, 0);
        }
        {
            const float* gp = dst + (size_t)(j0 + tid) * HEADS;
            float* lp = &dstT[buf][tid][0];
            __builtin_amdgcn_global_load_async_to_lds_b128(
                (gv4p)(uintptr_t)gp, (lv4p)(uint32_t)(uintptr_t)lp, 0, 0);
        }
#else
#pragma unroll
        for (int it = 0; it < 4; ++it) {
            const int r = r0 + it * 4;
            const float4 v = *reinterpret_cast<const float4*>(
                adj + (size_t)(i0 + r) * N_NODES + j0 + c);
            *reinterpret_cast<float4*>(&adjT[buf][r][c]) = v;
        }
        {
            const float4 dv =
                *reinterpret_cast<const float4*>(dst + (size_t)(j0 + tid) * HEADS);
            *reinterpret_cast<float4*>(&dstT[buf][tid][0]) = dv;
        }
#endif
    };

    v8f acc[4] = {v8f{}, v8f{}, v8f{}, v8f{}};

    stage(0, 0);  // prime the pipeline
    for (int k = 0; k < NB; ++k) {
        const int cur = k & 1;
#if GAT_ASYNC
        __builtin_amdgcn_s_wait_asynccnt(0);  // wave's copies into buf `cur` done
#endif
        __syncthreads();  // publish buf `cur`; everyone done reading buf `cur^1`
        if (k + 1 < NB) stage(cur ^ 1, (k + 1) * JB);  // overlap with compute

        const int j0 = k * JB;
        for (int sub = 0; sub < 4; ++sub) {  // K = 32 per subtile
            const int jb = sub * 32;
            // attention-prob A operand (f16) in contiguous K order, same as B:
            //  lanes 0-15 : e 0-15 -> K 0-15
            //  lanes 16-31: e 0-15 -> K 16-31
            v16h a;
#pragma unroll
            for (int e = 0; e < 16; ++e) {
                const int jj = jb + kb16 + e;
                const float ad = adjT[cur][mr][jj];
                float ev = srcv + dstT[cur][jj][head];
                ev = (ev > 0.f) ? ev : GAT_ALPHA * ev;
                const float p = (ad > 0.f) ? __expf(ev - mv) * isv : 0.f;
                a[e] = (_Float16)p;
            }
            const int jrow = j0 + jb + kb16;  // first B K-row for this lane
#pragma unroll
            for (int nt = 0; nt < 4; ++nt) {
                const _Float16* bp =
                    hT + (size_t)(head * HEAD_DIM + nt * 16 + mr) * N_NODES + jrow;
                const v16h b = *reinterpret_cast<const v16h*>(bp);
                acc[nt] = __builtin_amdgcn_wmma_f32_16x16x32_f16(
                    false, a, false, b, (short)0, acc[nt], false, false);
            }
        }
    }
    const int rbase = i0 + ((lane >> 4) << 3);
#pragma unroll
    for (int nt = 0; nt < 4; ++nt)
#pragma unroll
        for (int r = 0; r < 8; ++r)
            out[(size_t)(rbase + r) * OUT_FEAT + head * HEAD_DIM + nt * 16 + mr] =
                acc[nt][r];
}

// ---------------------------------------------------------------------------
// Kernel 6: in-place LayerNorm over 256 features, one block per row
// ---------------------------------------------------------------------------
__global__ __launch_bounds__(256) void gat_layernorm(float* __restrict__ out,
                                                     const float* __restrict__ gamma,
                                                     const float* __restrict__ beta) {
    __shared__ float red[256];
    const int i = blockIdx.x;
    const int t = threadIdx.x;
    const float v = out[(size_t)i * OUT_FEAT + t];

    red[t] = v;
    __syncthreads();
    for (int st = 128; st > 0; st >>= 1) {
        if (t < st) red[t] += red[t + st];
        __syncthreads();
    }
    const float mu = red[0] * (1.0f / OUT_FEAT);
    __syncthreads();
    const float dv = v - mu;
    red[t] = dv * dv;
    __syncthreads();
    for (int st = 128; st > 0; st >>= 1) {
        if (t < st) red[t] += red[t + st];
        __syncthreads();
    }
    const float var = red[0] * (1.0f / OUT_FEAT);
    const float rs = rsqrtf(var + LN_EPS);
    out[(size_t)i * OUT_FEAT + t] = dv * rs * gamma[t] + beta[t];
}

// ---------------------------------------------------------------------------
extern "C" void kernel_launch(void* const* d_in, const int* in_sizes, int n_in,
                              void* d_out, int out_size, void* d_ws, size_t ws_size,
                              hipStream_t stream) {
    const float* x = (const float*)d_in[0];      // [4096, 256]
    const float* adj = (const float*)d_in[1];    // [4096, 4096]
    const float* W = (const float*)d_in[2];      // [256, 256]
    const float* avec = (const float*)d_in[3];   // [128, 1]
    const float* gamma = (const float*)d_in[4];  // [256]
    const float* beta = (const float*)d_in[5];   // [256]
    float* out = (float*)d_out;                  // [4096, 256]

    float* ws = (float*)d_ws;
    float* h = ws;                                // 4096*256 f32
    float* src = h + (size_t)N_NODES * OUT_FEAT;  // 4096*4
    float* dst = src + N_NODES * HEADS;
    float* rowmax = dst + N_NODES * HEADS;
    float* rowsinv = rowmax + N_NODES * HEADS;
    _Float16* hT = (_Float16*)(rowsinv + N_NODES * HEADS);  // 256*4096 f16

    gat_gemm_xw<<<dim3(N_NODES / 16, 4), 128, 0, stream>>>(x, W, h);
    gat_srcdst<<<(N_NODES * HEADS + 255) / 256, 256, 0, stream>>>(h, avec, src, dst);
    gat_h2t<<<dim3(N_NODES / 32, OUT_FEAT / 32), 256, 0, stream>>>(h, hT);
    gat_rowstats<<<N_NODES, 256, 0, stream>>>(adj, src, dst, rowmax, rowsinv);
    gat_aggregate<<<N_NODES / 16, 128, 0, stream>>>(adj, hT, src, dst, rowmax,
                                                    rowsinv, out);
    gat_layernorm<<<N_NODES, 256, 0, stream>>>(out, gamma, beta);
}
